// feature_fusion_block_53506702573608
// MI455X (gfx1250) — matrix operand order repaired and checked by reading, not compiled, not run on hardware
//
#include <hip/hip_runtime.h>
#include <hip/hip_bf16.h>

typedef __attribute__((ext_vector_type(16))) _Float16 v16h;
typedef __attribute__((ext_vector_type(8)))  _Float16 v8h;
typedef __attribute__((ext_vector_type(8)))  float    v8f;

#define NT   139392   // B*H*W tokens
#define HWSZ 17424    // H*W
#define BB   8
#define HH   132
#define WW_  132

union FragH { v16h v; v8h h2[2]; };
union U8h   { v8h v; _Float16 h[8]; };

// ---------------- WMMA GEMM ----------------
// A: M x K row-major f16.  WT: N x K row-major f16 (i.e. W transposed).
// Workgroup = 256 thr (8 waves). Block tile 128(M) x 64(N); wave tile 32x32.
// flags bit0: exact GELU.  bit1: scatter f32 output as NCHW (C=256, HW=17424).

__device__ __forceinline__ v16h ffb_load_a(const _Float16* base, int ld, int lane) {
  // A 16x32 f16: lanes 0-15 row M=lane, K {0-7,16-23}; lanes 16-31 row M=lane-16, K {8-15,24-31}
  int r  = lane & 15;
  int kb = (lane < 16) ? 0 : 8;
  const _Float16* p = base + (size_t)r * ld + kb;
  FragH f;
  f.h2[0] = *(const v8h*)(p);
  f.h2[1] = *(const v8h*)(p + 16);
  return f.v;
}

__device__ __forceinline__ v16h ffb_load_b(const _Float16* base, int ld, int lane) {
  // B 32x16 f16: lanes 0-15 col N=lane, K {0-7,16-23}; lanes 16-31 col N=lane-16, K {8-15,24-31}
  int n  = lane & 15;
  int kb = (lane < 16) ? 0 : 8;
  const _Float16* p = base + (size_t)n * ld + kb;
  FragH f;
  f.h2[0] = *(const v8h*)(p);
  f.h2[1] = *(const v8h*)(p + 16);
  return f.v;
}

__device__ __forceinline__ void ffb_store_tile(v8f acc, int row0, int col0, int lane,
    const float* __restrict__ bias, _Float16* __restrict__ outH,
    float* __restrict__ outF, int N, int flags) {
  int n  = lane & 15;
  int mb = (lane < 16) ? 0 : 8;
  int col = col0 + n;
  float bv = bias[col];
#pragma unroll
  for (int r = 0; r < 8; ++r) {
    int row = row0 + mb + r;
    float v = acc[r] + bv;
    if (flags & 1) v = 0.5f * v * (1.0f + erff(v * 0.70710678118f));
    if (flags & 2) {
      int b  = row / HWSZ;
      int hw = row - b * HWSZ;
      outF[((size_t)b * 256 + col) * HWSZ + hw] = v;
    } else {
      outH[(size_t)row * N + col] = (_Float16)v;
    }
  }
}

__global__ __launch_bounds__(256) void ffb_gemm_kernel(
    const _Float16* __restrict__ A, const _Float16* __restrict__ WT,
    const float* __restrict__ bias, _Float16* __restrict__ outH,
    float* __restrict__ outF, int M, int N, int K, int flags) {
  int lane = threadIdx.x & 31;
  int wv   = threadIdx.x >> 5;
  int row0 = blockIdx.x * 128 + (wv & 3) * 32;
  int col0 = blockIdx.y * 64  + (wv >> 2) * 32;

  v8f c00 = {}, c01 = {}, c10 = {}, c11 = {};
  for (int k0 = 0; k0 < K; k0 += 32) {
    v16h a0 = ffb_load_a(A  + (size_t)row0        * K + k0, K, lane);
    v16h a1 = ffb_load_a(A  + (size_t)(row0 + 16) * K + k0, K, lane);
    v16h b0 = ffb_load_b(WT + (size_t)col0        * K + k0, K, lane);
    v16h b1 = ffb_load_b(WT + (size_t)(col0 + 16) * K + k0, K, lane);
    c00 = __builtin_amdgcn_wmma_f32_16x16x32_f16(false, a0, false, b0, (short)0, c00, false, false);
    c01 = __builtin_amdgcn_wmma_f32_16x16x32_f16(false, a0, false, b1, (short)0, c01, false, false);
    c10 = __builtin_amdgcn_wmma_f32_16x16x32_f16(false, a1, false, b0, (short)0, c10, false, false);
    c11 = __builtin_amdgcn_wmma_f32_16x16x32_f16(false, a1, false, b1, (short)0, c11, false, false);
  }
  ffb_store_tile(c00, row0,      col0,      lane, bias, outH, outF, N, flags);
  ffb_store_tile(c01, row0,      col0 + 16, lane, bias, outH, outF, N, flags);
  ffb_store_tile(c10, row0 + 16, col0,      lane, bias, outH, outF, N, flags);
  ffb_store_tile(c11, row0 + 16, col0 + 16, lane, bias, outH, outF, N, flags);
}

// ---------------- Weight transpose f32 (K,N) -> f16 (N,K) ----------------
__global__ void ffb_transpose_kernel(const float* __restrict__ W, _Float16* __restrict__ WT,
                                     int K, int N) {
  int i = blockIdx.x * blockDim.x + threadIdx.x;
  if (i >= K * N) return;
  int n = i / K;
  int k = i - n * K;
  WT[(size_t)n * K + k] = (_Float16)W[(size_t)k * N + n];
}

// ---------------- LayerNorm NCHW f32 -> NHWC f16 ----------------
// block 256 thr handles 16 consecutive hw tokens of one batch. grid.y picks x / x_e.
__global__ __launch_bounds__(256) void ffb_ln_kernel(
    const float* __restrict__ x, const float* __restrict__ xe,
    const float* __restrict__ nw, const float* __restrict__ nb,
    const float* __restrict__ new_, const float* __restrict__ neb,
    _Float16* __restrict__ xc, _Float16* __restrict__ xec) {
  __shared__ float tile[16 * 260];
  __shared__ float rsum[256];
  __shared__ float rsq[256];
  __shared__ float smean[16];
  __shared__ float srstd[16];

  const float* src; const float* w; const float* bi; _Float16* dst;
  if (blockIdx.y == 0) { src = x;  w = nw;   bi = nb;  dst = xc;  }
  else                 { src = xe; w = new_; bi = neb; dst = xec; }

  int blk = blockIdx.x;            // 0 .. 8711
  int b   = blk / 1089;
  int hw0 = (blk - b * 1089) * 16;
  int tk  = threadIdx.x & 15;
  int cg  = threadIdx.x >> 4;      // 0..15

  float s = 0.f, s2 = 0.f;
  for (int c = cg; c < 256; c += 16) {
    float v = src[((size_t)b * 256 + c) * HWSZ + hw0 + tk];
    tile[tk * 260 + c] = v;
    s += v; s2 += v * v;
  }
  rsum[cg * 16 + tk] = s;
  rsq [cg * 16 + tk] = s2;
  __syncthreads();
  if (threadIdx.x < 16) {
    float a = 0.f, a2 = 0.f;
    for (int g = 0; g < 16; ++g) { a += rsum[g * 16 + threadIdx.x]; a2 += rsq[g * 16 + threadIdx.x]; }
    float m = a * (1.f / 256.f);
    float var = a2 * (1.f / 256.f) - m * m;
    smean[threadIdx.x] = m;
    srstd[threadIdx.x] = rsqrtf(var + 1e-6f);
  }
  __syncthreads();
  int c = threadIdx.x;             // 0..255
  for (int t = 0; t < 16; ++t) {
    float v = tile[t * 260 + c];
    float y = (v - smean[t]) * srstd[t] * w[c] + bi[c];
    dst[((size_t)(b * HWSZ + hw0 + t)) * 256 + c] = (_Float16)y;
  }
}

// ---------------- 6x6 block-mean pooling of cat=[xc,xec] ----------------
// grid 384 (rows 288.. zero-padded), block 256.
__global__ __launch_bounds__(256) void ffb_pool_kernel(
    const _Float16* __restrict__ xc, const _Float16* __restrict__ xec,
    _Float16* __restrict__ pooled) {
  int pt = blockIdx.x;
  int c  = threadIdx.x;
  if (pt >= BB * 36) {
    pooled[(size_t)pt * 512 + c]       = (_Float16)0.f;
    pooled[(size_t)pt * 512 + 256 + c] = (_Float16)0.f;
    return;
  }
  int b = pt / 36, q = pt - b * 36;
  int qi = q / 6, qj = q - qi * 6;
  float s1 = 0.f, s2 = 0.f;
  for (int u = 0; u < 22; ++u)
    for (int v = 0; v < 22; ++v) {
      size_t t = (size_t)b * HWSZ + (qi * 22 + u) * WW_ + (qj * 22 + v);
      s1 += (float)xc [t * 256 + c];
      s2 += (float)xec[t * 256 + c];
    }
  pooled[(size_t)pt * 512 + c]       = (_Float16)(s1 * (1.f / 484.f));
  pooled[(size_t)pt * 512 + 256 + c] = (_Float16)(s2 * (1.f / 484.f));
}

// ---------------- attention: 36 queries vs 17424 keys per (b,head) ----------------
// grid 64 blocks (b*8+h), 8 waves; each wave does queries q = wave, wave+8, ...
__global__ __launch_bounds__(256) void ffb_attn_kernel(
    const _Float16* __restrict__ m_h, const _Float16* __restrict__ kv_h,
    float* __restrict__ g_small) {
  int bh = blockIdx.x;
  int b = bh >> 3, h = bh & 7;
  int wv = threadIdx.x >> 5;
  int lane = threadIdx.x & 31;

  for (int q = wv; q < 36; q += 8) {
    float mq[16];
    const _Float16* mp = m_h + ((size_t)(b * 36 + q)) * 128 + h * 16;
#pragma unroll
    for (int i = 0; i < 16; ++i) mq[i] = (float)mp[i] * 0.25f;   // d^-0.5, d=16

    float mi = -1e30f, li = 0.f;
    float acc[16];
#pragma unroll
    for (int i = 0; i < 16; ++i) acc[i] = 0.f;

    for (int t = lane; t < HWSZ; t += 32) {
      const _Float16* kp = kv_h + ((size_t)b * HWSZ + t) * 256 + h * 16;
      U8h k0, k1, v0, v1;
      k0.v = *(const v8h*)(kp);
      k1.v = *(const v8h*)(kp + 8);
      v0.v = *(const v8h*)(kp + 128);
      v1.v = *(const v8h*)(kp + 136);
      float s = 0.f;
#pragma unroll
      for (int i = 0; i < 8; ++i) s += mq[i] * (float)k0.h[i] + mq[8 + i] * (float)k1.h[i];
      float nm   = fmaxf(mi, s);
      float corr = __expf(mi - nm);
      float p    = __expf(s - nm);
      li = li * corr + p;
#pragma unroll
      for (int i = 0; i < 8; ++i) {
        acc[i]     = acc[i]     * corr + p * (float)v0.h[i];
        acc[8 + i] = acc[8 + i] * corr + p * (float)v1.h[i];
      }
      mi = nm;
    }
    // cross-lane online-softmax merge
    for (int off = 16; off >= 1; off >>= 1) {
      float mo = __shfl_xor(mi, off);
      float lo = __shfl_xor(li, off);
      float nm = fmaxf(mi, mo);
      float c1 = __expf(mi - nm);
      float c2 = __expf(mo - nm);
      li = li * c1 + lo * c2;
#pragma unroll
      for (int i = 0; i < 16; ++i) {
        float ao = __shfl_xor(acc[i], off);
        acc[i] = acc[i] * c1 + ao * c2;
      }
      mi = nm;
    }
    if (lane == 0) {
      float inv = 1.f / li;
#pragma unroll
      for (int i = 0; i < 16; ++i)
        g_small[((size_t)(b * 36 + q)) * 128 + h * 16 + i] = acc[i] * inv;
    }
  }
}

// ---------------- depthwise 7x7 conv (NHWC f16 -> f16), +bias ----------------
// grid (NT, 2): y picks (ef1->conv1) or (ef2->conv2). block 128 = channels.
__global__ __launch_bounds__(128) void ffb_dwconv_kernel(
    const _Float16* __restrict__ in1, const _Float16* __restrict__ in2,
    const float* __restrict__ wgt, const float* __restrict__ cbias,
    _Float16* __restrict__ out1, _Float16* __restrict__ out2) {
  const _Float16* in  = blockIdx.y ? in2  : in1;
  _Float16*       out = blockIdx.y ? out2 : out1;
  __shared__ float wl[128 * 49];
  for (int i = threadIdx.x; i < 128 * 49; i += 128) wl[i] = wgt[i];
  __syncthreads();
  int t = blockIdx.x;
  int c = threadIdx.x;
  int b  = t / HWSZ;
  int hw = t - b * HWSZ;
  int hh = hw / WW_;
  int ww = hw - hh * WW_;
  float acc = cbias[c];
  for (int dh = 0; dh < 7; ++dh) {
    int y = hh + dh - 3;
    if ((unsigned)y >= (unsigned)HH) continue;
    for (int dw = 0; dw < 7; ++dw) {
      int x2 = ww + dw - 3;
      if ((unsigned)x2 >= (unsigned)WW_) continue;
      acc += (float)in[((size_t)b * HWSZ + y * WW_ + x2) * 128 + c] * wl[c * 49 + dh * 7 + dw];
    }
  }
  out[(size_t)t * 128 + c] = (_Float16)acc;
}

// ---------------- combine: bilinear upsample g + q*e -> cat (NT x 384 f16) ----------------
// grid NT/2, block 256 (2 tokens x 128 channels)
__global__ __launch_bounds__(256) void ffb_combine_kernel(
    const float* __restrict__ g_small,
    const _Float16* __restrict__ q1, const _Float16* __restrict__ e1,
    const _Float16* __restrict__ q2, const _Float16* __restrict__ e2,
    _Float16* __restrict__ cat) {
  int idx = blockIdx.x * 2 + (threadIdx.x >> 7);
  int c   = threadIdx.x & 127;
  int b  = idx / HWSZ;
  int hw = idx - b * HWSZ;
  int hh = hw / WW_;
  int ww = hw - hh * WW_;
  float sh = (hh + 0.5f) * (1.f / 22.f) - 0.5f;
  float sw = (ww + 0.5f) * (1.f / 22.f) - 0.5f;
  int h0 = (int)floorf(sh); float fh = sh - (float)h0;
  int w0 = (int)floorf(sw); float fw = sw - (float)w0;
  int h0c = min(max(h0, 0), 5), h1c = min(max(h0 + 1, 0), 5);
  int w0c = min(max(w0, 0), 5), w1c = min(max(w0 + 1, 0), 5);
  const float* gb = g_small + (size_t)b * 36 * 128 + c;
  float g = (1.f - fh) * ((1.f - fw) * gb[(h0c * 6 + w0c) * 128] + fw * gb[(h0c * 6 + w1c) * 128])
          +        fh  * ((1.f - fw) * gb[(h1c * 6 + w0c) * 128] + fw * gb[(h1c * 6 + w1c) * 128]);
  size_t t = (size_t)idx;
  cat[t * 384 + c]       = (_Float16)g;
  cat[t * 384 + 128 + c] = (_Float16)((float)q1[t * 128 + c] * (float)e1[t * 128 + c]);
  cat[t * 384 + 256 + c] = (_Float16)((float)q2[t * 128 + c] * (float)e2[t * 128 + c]);
}

// ---------------- host launcher ----------------
extern "C" void kernel_launch(void* const* d_in, const int* in_sizes, int n_in,
                              void* d_out, int out_size, void* d_ws, size_t ws_size,
                              hipStream_t stream) {
  const float* x      = (const float*)d_in[0];
  const float* x_e    = (const float*)d_in[1];
  const float* norm_w = (const float*)d_in[2];
  const float* norm_b = (const float*)d_in[3];
  const float* norme_w= (const float*)d_in[4];
  const float* norme_b= (const float*)d_in[5];
  const float* l_w    = (const float*)d_in[6];
  const float* l_b    = (const float*)d_in[7];
  const float* kv_w   = (const float*)d_in[8];
  const float* kv_b   = (const float*)d_in[9];
  const float* xe_w   = (const float*)d_in[10];
  const float* xe_b   = (const float*)d_in[11];
  const float* q_w    = (const float*)d_in[12];
  const float* q_b    = (const float*)d_in[13];
  const float* ef_w   = (const float*)d_in[14];
  const float* ef_b   = (const float*)d_in[15];
  const float* ec_w   = (const float*)d_in[16];
  const float* ec_b   = (const float*)d_in[17];
  const float* eb_w   = (const float*)d_in[18];
  const float* eb_b   = (const float*)d_in[19];
  const float* proj_w = (const float*)d_in[20];
  const float* proj_b = (const float*)d_in[21];
  const float* proje_w= (const float*)d_in[22];
  const float* proje_b= (const float*)d_in[23];

  char* ws = (char*)d_ws;
  const size_t SZ256 = (size_t)NT * 256 * 2;   // bytes of one 256-ch f16 tensor
  const size_t SZ128 = (size_t)NT * 128 * 2;

  size_t oXC   = 0;                 // xc_h ; later conv1
  size_t oXEC  = oXC  + SZ256;      // xec_h; later conv2
  size_t oC1   = oXEC + SZ256;      // xg   ; later q1 @ oC1, q2 @ oC1+SZ128
  size_t oC2   = oC1  + SZ256;      // kv   ; later ef1/ef2 ; later e1/e2
  size_t oCAT  = oC2  + SZ256;      // cat (NT x 384 f16)
  size_t oPOOL = oCAT + (size_t)NT * 384 * 2;
  size_t oM    = oPOOL + 384 * 512 * 2;
  size_t oG    = oM    + 384 * 128 * 2;
  size_t oW    = oG    + (size_t)BB * 36 * 128 * 4;
  size_t oLW   = oW;
  size_t oKVW  = oLW  + 256 * 256 * 2;
  size_t oXEW  = oKVW + 256 * 256 * 2;
  size_t oQW   = oXEW + 128 * 512 * 2;
  size_t oEFW  = oQW  + 128 * 256 * 2;
  size_t oEBW  = oEFW + 128 * 256 * 2;
  size_t oPW   = oEBW + 128 * 128 * 2;
  size_t oPEW  = oPW  + 256 * 384 * 2;

  _Float16* xc_h   = (_Float16*)(ws + oXC);
  _Float16* xec_h  = (_Float16*)(ws + oXEC);
  _Float16* xg_h   = (_Float16*)(ws + oC1);
  _Float16* q1_h   = (_Float16*)(ws + oC1);
  _Float16* q2_h   = (_Float16*)(ws + oC1 + SZ128);
  _Float16* kv_h   = (_Float16*)(ws + oC2);
  _Float16* ef1_h  = (_Float16*)(ws + oC2);
  _Float16* ef2_h  = (_Float16*)(ws + oC2 + SZ128);
  _Float16* e1_h   = (_Float16*)(ws + oC2);
  _Float16* e2_h   = (_Float16*)(ws + oC2 + SZ128);
  _Float16* conv1_h= (_Float16*)(ws + oXC);
  _Float16* conv2_h= (_Float16*)(ws + oXEC);
  _Float16* cat_h  = (_Float16*)(ws + oCAT);
  _Float16* pooled = (_Float16*)(ws + oPOOL);
  _Float16* m_h    = (_Float16*)(ws + oM);
  float*    g_sm   = (float*)   (ws + oG);
  _Float16* l_wT   = (_Float16*)(ws + oLW);
  _Float16* kv_wT  = (_Float16*)(ws + oKVW);
  _Float16* xe_wT  = (_Float16*)(ws + oXEW);
  _Float16* q_wT   = (_Float16*)(ws + oQW);
  _Float16* ef_wT  = (_Float16*)(ws + oEFW);
  _Float16* eb_wT  = (_Float16*)(ws + oEBW);
  _Float16* proj_wT= (_Float16*)(ws + oPW);
  _Float16* proje_wT=(_Float16*)(ws + oPEW);

  float* out_f  = (float*)d_out;
  float* oute_f = out_f + (size_t)NT * 256;

  // 1) weight transposes (f32 (K,N) -> f16 (N,K))
  auto tr = [&](const float* W, _Float16* WT, int K, int N) {
    int tot = K * N;
    ffb_transpose_kernel<<<(tot + 255) / 256, 256, 0, stream>>>(W, WT, K, N);
  };
  tr(l_w,    l_wT,    256, 256);
  tr(kv_w,   kv_wT,   256, 256);
  tr(xe_w,   xe_wT,   512, 128);
  tr(q_w,    q_wT,    256, 128);
  tr(ef_w,   ef_wT,   256, 128);
  tr(eb_w,   eb_wT,   128, 128);
  tr(proj_w, proj_wT, 384, 256);
  tr(proje_w,proje_wT,384, 256);

  // 2) LayerNorm both inputs -> NHWC f16
  ffb_ln_kernel<<<dim3(1089 * BB, 2), 256, 0, stream>>>(
      x, x_e, norm_w, norm_b, norme_w, norme_b, xc_h, xec_h);

  // 3) pooling -> padded 384x512
  ffb_pool_kernel<<<384, 256, 0, stream>>>(xc_h, xec_h, pooled);

  // 4) m = pooled @ xe_w + xe_b   (384 x 512 x 128)
  ffb_gemm_kernel<<<dim3(3, 2), 256, 0, stream>>>(pooled, xe_wT, xe_b, m_h, nullptr,
                                                  384, 128, 512, 0);
  // 5) xg = gelu(xc @ l_w + l_b)  (NT x 256 x 256)
  ffb_gemm_kernel<<<dim3(1089, 4), 256, 0, stream>>>(xc_h, l_wT, l_b, xg_h, nullptr,
                                                     NT, 256, 256, 1);
  // 6) kv = xg @ kv_w + kv_b
  ffb_gemm_kernel<<<dim3(1089, 4), 256, 0, stream>>>(xg_h, kv_wT, kv_b, kv_h, nullptr,
                                                     NT, 256, 256, 0);
  // 7) attention -> g_small
  ffb_attn_kernel<<<64, 256, 0, stream>>>(m_h, kv_h, g_sm);

  // 8) q1 = xc @ q_w ; q2 = xec @ q_w   (overwrites xg region)
  ffb_gemm_kernel<<<dim3(1089, 2), 256, 0, stream>>>(xc_h,  q_wT, q_b, q1_h, nullptr,
                                                     NT, 128, 256, 0);
  ffb_gemm_kernel<<<dim3(1089, 2), 256, 0, stream>>>(xec_h, q_wT, q_b, q2_h, nullptr,
                                                     NT, 128, 256, 0);
  // 9) ef1 = xec @ ef_w ; ef2 = xc @ ef_w   (overwrites kv region; attn already done)
  ffb_gemm_kernel<<<dim3(1089, 2), 256, 0, stream>>>(xec_h, ef_wT, ef_b, ef1_h, nullptr,
                                                     NT, 128, 256, 0);
  ffb_gemm_kernel<<<dim3(1089, 2), 256, 0, stream>>>(xc_h,  ef_wT, ef_b, ef2_h, nullptr,
                                                     NT, 128, 256, 0);
  // 10) depthwise conv (overwrites xc/xec regions; both dead now)
  ffb_dwconv_kernel<<<dim3(NT, 2), 128, 0, stream>>>(ef1_h, ef2_h, ec_w, ec_b,
                                                     conv1_h, conv2_h);
  // 11) e = conv @ eb_w + eb_b  (overwrites ef regions)
  ffb_gemm_kernel<<<dim3(1089, 2), 256, 0, stream>>>(conv1_h, eb_wT, eb_b, e1_h, nullptr,
                                                     NT, 128, 128, 0);
  ffb_gemm_kernel<<<dim3(1089, 2), 256, 0, stream>>>(conv2_h, eb_wT, eb_b, e2_h, nullptr,
                                                     NT, 128, 128, 0);
  // 12) combine: upsample g + q*e -> cat
  ffb_combine_kernel<<<NT / 2, 256, 0, stream>>>(g_sm, q1_h, e1_h, q2_h, e2_h, cat_h);

  // 13) out = cat @ proj_w + proj_b (NCHW f32); out_e = cat @ proje_w + proje_b
  ffb_gemm_kernel<<<dim3(1089, 4), 256, 0, stream>>>(cat_h, proj_wT,  proj_b,  nullptr,
                                                     out_f,  NT, 256, 384, 2);
  ffb_gemm_kernel<<<dim3(1089, 4), 256, 0, stream>>>(cat_h, proje_wT, proje_b, nullptr,
                                                     oute_f, NT, 256, 384, 2);
}
